// CgpHmmLayer_55087250538514
// MI455X (gfx1250) — compile-verified, matrix operand order checked by reading, not run on hardware
//
#include <hip/hip_runtime.h>

typedef __attribute__((ext_vector_type(16))) __bf16  v16bf;
typedef __attribute__((ext_vector_type(8)))  float   v8f;

constexpr int T_LEN = 2048, S_N = 512, E_N = 6;
constexpr int NT = S_N / 16;     // 32 n-tiles of 16 columns
constexpr int KB = S_N / 32;     // 16 k-blocks of depth 32
constexpr int KB_LDS = 8;        // k-blocks of A cached in LDS (256 KB)

// Position of element (k-within-block kq, 16-lane index ln16) inside a
// 512-element bf16 WMMA fragment, per CDNA5 ISA 7.12.2 16-bit layout.
__device__ __forceinline__ int frag_pos(int kq, int ln16) {
  int hh = (kq & 15) >> 3;                  // lane half that owns this k
  int e  = ((kq >> 4) << 3) | (kq & 7);     // element 0..15 within the lane
  return (ln16 + 16 * hh) * 16 + e;
}

__device__ __forceinline__ unsigned short f2bf(float x) {  // RNE f32 -> bf16
  unsigned u = __float_as_uint(x);
  u += 0x7FFFu + ((u >> 16) & 1u);
  return (unsigned short)(u >> 16);
}

// ---- one-time: row softmax of A_kernel, emit bf16 in WMMA-B fragment order ----
__global__ __launch_bounds__(256) void softmaxA_swz(const float* __restrict__ Ak,
                                                    unsigned short* __restrict__ Aswz) {
  __shared__ float red[256];
  const int k = blockIdx.x;          // K row (source state)
  const int tid = threadIdx.x;
  float a = Ak[k * S_N + tid];
  float b = Ak[k * S_N + tid + 256];
  red[tid] = fmaxf(a, b);
  __syncthreads();
  for (int s = 128; s > 0; s >>= 1) { if (tid < s) red[tid] = fmaxf(red[tid], red[tid + s]); __syncthreads(); }
  const float mx = red[0];
  __syncthreads();
  float ea = expf(a - mx), eb = expf(b - mx);
  red[tid] = ea + eb;
  __syncthreads();
  for (int s = 128; s > 0; s >>= 1) { if (tid < s) red[tid] += red[tid + s]; __syncthreads(); }
  const float inv = 1.0f / red[0];
  const int kb = k >> 5, kq = k & 31;
  int j = tid;
  Aswz[(kb * NT + (j >> 4)) * 512 + frag_pos(kq, j & 15)] = f2bf(ea * inv);
  j = tid + 256;
  Aswz[(kb * NT + (j >> 4)) * 512 + frag_pos(kq, j & 15)] = f2bf(eb * inv);
}

// ---- one-time: softmax of B_kernel rows (f32) and of I_kernel ----
__global__ __launch_bounds__(256) void softmaxBI(const float* __restrict__ Bk,
                                                 const float* __restrict__ Ik,
                                                 float* __restrict__ Bsm,
                                                 float* __restrict__ Ism) {
  const int tid = threadIdx.x;
  if (blockIdx.x < 2) {
    int r = blockIdx.x * 256 + tid;
    float v[E_N], mx = -3.4e38f;
    for (int e = 0; e < E_N; ++e) { v[e] = Bk[r * E_N + e]; mx = fmaxf(mx, v[e]); }
    float s = 0.f;
    for (int e = 0; e < E_N; ++e) { v[e] = expf(v[e] - mx); s += v[e]; }
    float inv = 1.0f / s;
    for (int e = 0; e < E_N; ++e) Bsm[r * E_N + e] = v[e] * inv;
  } else {
    __shared__ float red[256];
    float a = Ik[tid], b = Ik[tid + 256];
    red[tid] = fmaxf(a, b);
    __syncthreads();
    for (int s = 128; s > 0; s >>= 1) { if (tid < s) red[tid] = fmaxf(red[tid], red[tid + s]); __syncthreads(); }
    float mx = red[0];
    __syncthreads();
    float ea = expf(a - mx), eb = expf(b - mx);
    red[tid] = ea + eb;
    __syncthreads();
    for (int s = 128; s > 0; s >>= 1) { if (tid < s) red[tid] += red[tid + s]; __syncthreads(); }
    float inv = 1.0f / red[0];
    Ism[tid] = ea * inv;
    Ism[tid + 256] = eb * inv;
  }
}

// ---- main scan: 4 WGs, each owns 16 batch rows for all 2047 steps ----
__global__ __launch_bounds__(256, 1) void hmm_forward(
    const int* __restrict__ obs,
    const unsigned short* __restrict__ Aswz,
    const float* __restrict__ Bsm,
    const float* __restrict__ Ism,
    float* __restrict__ out)
{
  __shared__ __align__(64) unsigned short ldsB[KB_LDS * NT * 512];  // 256 KB A-frags
  __shared__ __align__(64) unsigned short alphaFrag[KB * 512];      // 16 KB alpha frags
  __shared__ float BmLDS[S_N * E_N];                                // 12 KB emissions
  __shared__ float zpart[8][16];
  __shared__ float invz[16];
  __shared__ float z0s[16];
  __shared__ int   obsLDS[16];

  const int tid  = threadIdx.x;
  const int wv   = tid >> 5;          // wave 0..7
  const int lane = tid & 31;
  const int hh   = lane >> 4;         // lane half: rows 0-7 vs 8-15 of C/D
  const int ln16 = lane & 15;
  const int b0   = blockIdx.x * 16;   // this WG's batch rows
  const int nt0  = wv * 4;            // this wave's 4 n-tiles (64 columns)

  // stage: emission table + first half of swizzled A into LDS
  for (int i = tid; i < S_N * E_N; i += 256) BmLDS[i] = Bsm[i];
  {
    const uint4* src = (const uint4*)Aswz;
    uint4* dst = (uint4*)ldsB;
    const int n16 = (KB_LDS * NT * 512) / 8;   // uint4 = 8 ushorts
    for (int i = tid; i < n16; i += 256) dst[i] = src[i];
  }
  if (tid < 16) { obsLDS[tid] = obs[(b0 + tid) * T_LEN]; z0s[tid] = 0.f; }
  __syncthreads();

  // t = 0: alpha0 = I * em0, scale, log
  const int m0 = tid >> 4;
  const int sb = (tid & 15) * 32;
  float v0[32];
  {
    float ls = 0.f;
    const int e0 = obsLDS[m0];
    for (int i = 0; i < 32; ++i) {
      int s = sb + i;
      float val = Ism[s] * BmLDS[s * E_N + e0];
      v0[i] = val; ls += val;
    }
    atomicAdd(&z0s[m0], ls);
  }
  __syncthreads();
  float loglik = 0.f;                  // owned by tid<16 (row = tid)
  if (tid < 16) { float z = z0s[tid]; invz[tid] = 1.0f / z; loglik = logf(z); }
  __syncthreads();
  {
    const float inv = invz[m0];
    for (int i = 0; i < 32; ++i) {
      int s = sb + i;
      alphaFrag[(s >> 5) * 512 + frag_pos(s & 31, m0)] = f2bf(v0[i] * inv);
    }
  }
  __syncthreads();

  const v16bf* __restrict__ gB = (const v16bf*)Aswz;
  const unsigned short* afBase = &alphaFrag[lane * 16];

  for (int t = 1; t < T_LEN; ++t) {
    if (tid < 16) obsLDS[tid] = obs[(b0 + tid) * T_LEN + t];

    v8f acc[4] = {};

    // prologue: kick off first streamed k-block's global loads
    v16bf nb0, nb1, nb2, nb3;
    {
      const v16bf* g = gB + (KB_LDS * NT + nt0) * 32 + lane;
      nb0 = g[0]; nb1 = g[32]; nb2 = g[64]; nb3 = g[96];
    }

    // Interleave one LDS-cached k-block (i) with one L2-streamed k-block
    // (KB_LDS+i) per iteration so the DS and VMEM pipes run concurrently.
    #pragma unroll 2
    for (int i = 0; i < KB_LDS; ++i) {
      v16bf afL = *(const v16bf*)(afBase + i * 512);
      const v16bf* lb = (const v16bf*)(&ldsB[(i * NT + nt0) * 512]) + lane;
      v16bf l0 = lb[0];
      v16bf l1 = lb[32];
      v16bf l2 = lb[64];
      v16bf l3 = lb[96];
      v16bf afG = *(const v16bf*)(afBase + (KB_LDS + i) * 512);
      v16bf g0 = nb0, g1 = nb1, g2 = nb2, g3 = nb3;
      if (i + 1 < KB_LDS) {   // start next streamed block's loads (double buffer)
        const v16bf* g = gB + ((KB_LDS + i + 1) * NT + nt0) * 32 + lane;
        nb0 = g[0]; nb1 = g[32]; nb2 = g[64]; nb3 = g[96];
      }
      acc[0] = __builtin_amdgcn_wmma_f32_16x16x32_bf16(false, afL, false, l0, (short)0, acc[0], false, false);
      acc[1] = __builtin_amdgcn_wmma_f32_16x16x32_bf16(false, afL, false, l1, (short)0, acc[1], false, false);
      acc[2] = __builtin_amdgcn_wmma_f32_16x16x32_bf16(false, afL, false, l2, (short)0, acc[2], false, false);
      acc[3] = __builtin_amdgcn_wmma_f32_16x16x32_bf16(false, afL, false, l3, (short)0, acc[3], false, false);
      acc[0] = __builtin_amdgcn_wmma_f32_16x16x32_bf16(false, afG, false, g0, (short)0, acc[0], false, false);
      acc[1] = __builtin_amdgcn_wmma_f32_16x16x32_bf16(false, afG, false, g1, (short)0, acc[1], false, false);
      acc[2] = __builtin_amdgcn_wmma_f32_16x16x32_bf16(false, afG, false, g2, (short)0, acc[2], false, false);
      acc[3] = __builtin_amdgcn_wmma_f32_16x16x32_bf16(false, afG, false, g3, (short)0, acc[3], false, false);
    }
    __syncthreads();   // all alphaFrag reads done; obsLDS visible

    // emission multiply + per-row sums (C/D layout: lane half hh holds rows r+8*hh)
    float rs[8];
    #pragma unroll
    for (int r = 0; r < 8; ++r) rs[r] = 0.f;
    #pragma unroll
    for (int ti = 0; ti < 4; ++ti) {
      const int s = (nt0 + ti) * 16 + ln16;
      #pragma unroll
      for (int r = 0; r < 8; ++r) {
        const int m = r + 8 * hh;
        float val = acc[ti][r] * BmLDS[s * E_N + obsLDS[m]];
        acc[ti][r] = val;
        rs[r] += val;
      }
    }
    #pragma unroll
    for (int r = 0; r < 8; ++r) {       // butterfly over each 16-lane half
      float v = rs[r];
      v += __shfl_xor(v, 1, 32);
      v += __shfl_xor(v, 2, 32);
      v += __shfl_xor(v, 4, 32);
      v += __shfl_xor(v, 8, 32);
      rs[r] = v;
    }
    if (ln16 == 0) {
      #pragma unroll
      for (int r = 0; r < 8; ++r) zpart[wv][r + 8 * hh] = rs[r];
    }
    __syncthreads();
    if (tid < 16) {
      float z = 0.f;
      #pragma unroll
      for (int w = 0; w < 8; ++w) z += zpart[w][tid];
      loglik += logf(z);
      invz[tid] = 1.0f / z;
    }
    __syncthreads();
    // normalize, convert to bf16, scatter into next step's A-operand fragments
    #pragma unroll
    for (int ti = 0; ti < 4; ++ti) {
      const int n = (nt0 + ti) * 16 + ln16;
      const int base = (n >> 5) * 512;
      const int kq = n & 31;
      #pragma unroll
      for (int r = 0; r < 8; ++r) {
        const int m = r + 8 * hh;
        alphaFrag[base + frag_pos(kq, m)] = f2bf(acc[ti][r] * invz[m]);
      }
    }
    __syncthreads();
  }

  if (tid < 16) out[b0 + tid] = loglik;
}

extern "C" void kernel_launch(void* const* d_in, const int* in_sizes, int n_in,
                              void* d_out, int out_size, void* d_ws, size_t ws_size,
                              hipStream_t stream) {
  (void)in_sizes; (void)n_in; (void)out_size; (void)ws_size;
  const int*   obs = (const int*)d_in[0];
  const float* Ak  = (const float*)d_in[1];
  const float* Bk  = (const float*)d_in[2];
  const float* Ik  = (const float*)d_in[3];
  float* out = (float*)d_out;

  unsigned short* Aswz = (unsigned short*)d_ws;                       // 512 KB
  float* Bsm = (float*)((char*)d_ws + S_N * S_N * sizeof(unsigned short));
  float* Ism = Bsm + S_N * E_N;

  softmaxA_swz<<<S_N, 256, 0, stream>>>(Ak, Aswz);
  softmaxBI<<<3, 256, 0, stream>>>(Bk, Ik, Bsm, Ism);
  hmm_forward<<<4, 256, 0, stream>>>(obs, Aswz, Bsm, Ism, out);
}
